// SpectralConvolutionBlock_17033840296147
// MI455X (gfx1250) — matrix operand order repaired and checked by reading, not compiled, not run on hardware
//
#include <hip/hip_runtime.h>
#include <hip/hip_bf16.h>
#include <math.h>

// ---------------------------------------------------------------------------
// MI455X (gfx1250) implementation of SpectralConvolutionBlock.
// Memory-bound (~27 GFLOP vs ~300 MB @ 23.3 TB/s) -> bf16 streamed operands,
// f32 WMMA accumulation via v_wmma_f32_16x16x32_bf16 (wave32).
// GEMM inner loop: double-buffered LDS, A tile staged with
// global_load_async_to_lds_b64 (ASYNCcnt), B tile staged transposed so all
// WMMA fragments load as ds_load_b128 pairs.
// ---------------------------------------------------------------------------

typedef __bf16 bf16;
typedef __attribute__((ext_vector_type(16))) __bf16 v16bf;
typedef __attribute__((ext_vector_type(8)))  float  v8f;

#define TWO_PI 6.28318530717958647692f
#define USE_ASYNC_LDS 1

__device__ __forceinline__ float gelu_exact(float x) {
    return 0.5f * x * (1.0f + erff(x * 0.70710678118654752f));
}

__device__ __forceinline__ void async_copy_b64(void* lds_dst, const void* gsrc) {
#if USE_ASYNC_LDS
    unsigned ldsa = (unsigned)(unsigned long long)lds_dst;          // low 32b = LDS offset
    unsigned long long ga = (unsigned long long)gsrc;
    asm volatile("global_load_async_to_lds_b64 %0, %1, off"
                 :: "v"(ldsa), "v"(ga) : "memory");
#else
    *(uint2*)lds_dst = *(const uint2*)gsrc;
#endif
}

__device__ __forceinline__ void wait_async_lds() {
#if USE_ASYNC_LDS
    asm volatile("s_wait_asynccnt 0" ::: "memory");
#endif
}

union FragBF { v16bf v; uint4 q[2]; };
union Pack4  { bf16 e[4]; uint2 q; };

// ---------------------------- utility kernels ------------------------------

// f32 -> bf16, 4 elements per thread (n % 4 == 0)
__global__ void k_cvt_bf16_v4(const float* __restrict__ in, bf16* __restrict__ out, int n4) {
    int i = blockIdx.x * blockDim.x + threadIdx.x;
    if (i < n4) {
        float4 v = *(const float4*)(in + 4 * (long long)i);
        Pack4 p;
        p.e[0] = (bf16)v.x; p.e[1] = (bf16)v.y; p.e[2] = (bf16)v.z; p.e[3] = (bf16)v.w;
        *(uint2*)(out + 4 * (long long)i) = p.q;
    }
}

// Tiled transpose + convert: out[z][c][r] = (bf16)in[z][r][c].
// grid = (C/32, R/32, batch), block = 256. Coalesced reads and 8B writes.
__global__ void __launch_bounds__(256)
k_transpose_cvt_bf16(const float* __restrict__ in, bf16* __restrict__ out, int R, int C) {
    __shared__ float T[32][33];
    const int z = blockIdx.z;
    const float* inz = in + (long long)z * R * C;
    bf16* outz = out + (long long)z * R * C;
    const int r0 = blockIdx.y * 32, c0 = blockIdx.x * 32;
    const int tid = threadIdx.x;
    {
        int r = tid >> 3, c4 = (tid & 7) << 2;
        float4 v = *(const float4*)(inz + (long long)(r0 + r) * C + c0 + c4);
        T[r][c4 + 0] = v.x; T[r][c4 + 1] = v.y; T[r][c4 + 2] = v.z; T[r][c4 + 3] = v.w;
    }
    __syncthreads();
    {
        int c = tid >> 3, r4 = (tid & 7) << 2;
        Pack4 p;
        p.e[0] = (bf16)T[r4 + 0][c];
        p.e[1] = (bf16)T[r4 + 1][c];
        p.e[2] = (bf16)T[r4 + 2][c];
        p.e[3] = (bf16)T[r4 + 3][c];
        *(uint2*)(outz + (long long)(c0 + c) * R + r0 + r4) = p.q;
    }
}

// Forward DFT basis, [128][8192]: rows 0..63 = cos(2*pi*m*s/8192),
// rows 64..127 = -sin(2*pi*m*s/8192)   (ut = F @ u, e^{-i t} = cos - i sin)
__global__ void k_gen_fwd_basis(bf16* __restrict__ out) {
    int i = blockIdx.x * blockDim.x + threadIdx.x;   // 128*8192 elements
    int s = i & 8191, row = i >> 13, m = row & 63;
    int t = (m * s) & 8191;                          // exact range reduction
    float th = (float)t * (TWO_PI / 8192.0f);
    float sv = __sinf(th), cv = __cosf(th);
    out[i] = (bf16)((row < 64) ? cv : -sv);
}

// Inverse DFT basis, [8192][128]: col j<64 -> cos/8192, col j>=64 -> -sin/8192
__global__ void k_gen_inv_basis(bf16* __restrict__ out) {
    int i = blockIdx.x * blockDim.x + threadIdx.x;   // 8192*128 elements
    int j = i & 127, s = i >> 7, m = j & 63;
    int t = (m * s) & 8191;
    float th = (float)t * (TWO_PI / 8192.0f);
    float sv = __sinf(th), cv = __cosf(th);
    out[i] = (bf16)(((j < 64) ? cv : -sv) * (1.0f / 8192.0f));
}

// 64-point circular convolution along the mode axis (elementwise in h):
// conv[b,n,h] = (1/8) * sum_m ut[b,m,h] * s[b,(n-m) mod 64, h]   (complex)
__global__ void k_circconv(const float* __restrict__ ut,
                           const float* __restrict__ s_r,
                           const float* __restrict__ s_i,
                           bf16* __restrict__ conv_bf) {
    int idx = blockIdx.x * blockDim.x + threadIdx.x;   // 8*64*256 = 131072
    int h = idx & 255;
    int n = (idx >> 8) & 63;
    int b = idx >> 14;
    const float* utb = ut  + (long long)b * (128 * 256);
    const float* srb = s_r + (long long)b * (64 * 256);
    const float* sib = s_i + (long long)b * (64 * 256);
    float cr = 0.0f, ci = 0.0f;
    #pragma unroll 4
    for (int m = 0; m < 64; ++m) {
        float ur = utb[m * 256 + h];
        float ui = utb[(64 + m) * 256 + h];
        int j = (n - m) & 63;
        float vr = srb[j * 256 + h];
        float vi = sib[j * 256 + h];
        cr += ur * vr - ui * vi;
        ci += ur * vi + ui * vr;
    }
    cr *= 0.125f; ci *= 0.125f;
    conv_bf[(long long)b * (128 * 256) + n * 256 + h]        = (bf16)cr;
    conv_bf[(long long)b * (128 * 256) + (64 + n) * 256 + h] = (bf16)ci;
}

// ---------------------- generic bf16 WMMA GEMM -----------------------------
// C[M x N] = act( A[M x K] @ B[K x N] + addend + bias ), batched over gridDim.z.
// Block = 256 threads = 8 wave32, 2x4 waves of 16x16 tiles -> 32x64 block tile.
// M % 32 == 0, N % 64 == 0, K % 32 == 0 (all shapes here satisfy this).

template <bool TADD, bool TBIAS, bool TGELU, bool TOUTF, bool TOUTB>
__global__ void __launch_bounds__(256)
k_gemm_bf16(const bf16* __restrict__ A, const bf16* __restrict__ B,
            float* __restrict__ Cf, bf16* __restrict__ Cb,
            const bf16* __restrict__ addend, const float* __restrict__ bias,
            int K, int lda, int ldb, int ldc,
            long long sA, long long sB, long long sC) {
    __shared__ __align__(16) bf16 As[2][32][40];    // [buf][row][k]   (+8 pad)
    __shared__ __align__(16) bf16 BsT[2][64][40];   // [buf][col][k]   transposed

    const int z = blockIdx.z;
    const bf16* Az = A + (long long)z * sA;
    const bf16* Bz = B + (long long)z * sB;
    const long long cbase = (long long)z * sC;

    const int row0 = blockIdx.y * 32;
    const int col0 = blockIdx.x * 64;
    const int tid  = threadIdx.x;
    const int lane = tid & 31;
    const int wid  = tid >> 5;
    const int wrow = (wid >> 2) << 4;   // 0 or 16
    const int wcol = (wid & 3) << 4;    // 0,16,32,48

    const int sr = tid >> 3;            // staging row 0..31
    const int ca = (tid & 7) << 2;      // A staging col (4 bf16 = 8B / thread)
    const int cb = (tid & 7) << 3;      // B staging col (8 bf16 = 16B / thread)

    const int mrow = lane & 15;
    const int hl   = lane >> 4;
    const int ncol = lane & 15;

    const bf16* aRow = Az + (long long)(row0 + sr) * lda + ca;  // + k0
    v8f acc = {};

    auto stage = [&](int bfi, int k0) {
        async_copy_b64(&As[bfi][sr][ca], aRow + k0);
        union { uint4 q; bf16 e[8]; } t;
        t.q = *(const uint4*)(Bz + (long long)(k0 + sr) * ldb + col0 + cb);
        #pragma unroll
        for (int j = 0; j < 8; ++j) BsT[bfi][cb + j][sr] = t.e[j];
    };

    stage(0, 0);
    wait_async_lds();
    __syncthreads();

    int buf = 0;
    for (int k0 = 0; k0 < K; k0 += 32) {
        if (k0 + 32 < K) stage(buf ^ 1, k0 + 32);

        // A fragment (ISA 7.12.2, 16-bit A 16x32): per lane two contiguous
        // 8-element runs at k = half*8 and k = 16 + half*8.
        FragBF a;
        const bf16* ap = &As[buf][wrow + mrow][hl << 3];
        a.q[0] = *(const uint4*)(ap);
        a.q[1] = *(const uint4*)(ap + 16);
        // B fragment (32x16): per lane two 8-element k runs at k = half*8 and k = 16 + half*8.
        FragBF b;
        const bf16* bp = &BsT[buf][wcol + ncol][hl << 3];
        b.q[0] = *(const uint4*)(bp);
        b.q[1] = *(const uint4*)(bp + 16);

        acc = __builtin_amdgcn_wmma_f32_16x16x32_bf16(false, a.v, false, b.v,
                                                      (short)0, acc, false, false);
        wait_async_lds();
        __syncthreads();
        buf ^= 1;
    }

    // C/D layout: VGPR r holds row (r + 8*half), col = lane&15
    const int gcol = col0 + wcol + ncol;
    #pragma unroll
    for (int r = 0; r < 8; ++r) {
        int grow = row0 + wrow + r + (hl << 3);
        long long o = cbase + (long long)grow * ldc + gcol;
        float x = acc[r];
        if (TADD)  x += (float)addend[o];
        if (TBIAS) x += bias[gcol];
        if (TGELU) x = gelu_exact(x);
        if (TOUTF) Cf[o] = x;
        if (TOUTB) Cb[o] = (bf16)x;
    }
}

// ------------------------- per-mode complex mix ----------------------------
// For each mode m: Wu = ut[:,m,:] @ (Wr[m] + i Wi[m]), then s = gelu(Wu.re/im).
// Weights are pre-transposed per mode (WrT[m][n][k] = Wr[m][k][n]) so every
// B fragment is two contiguous 16B LDS loads; A fragments come straight from
// ut (k contiguous). A rows are the 8 batches (rows 8..15 zero-padded).
__global__ void __launch_bounds__(256)
k_modemix(const bf16* __restrict__ ut,    // [8][128][256] (rows 0..63 re, 64..127 im)
          const bf16* __restrict__ WrT,   // [64][256][256], [m][n][k]
          const bf16* __restrict__ WiT,
          float* __restrict__ s_r,        // [8][64][256]
          float* __restrict__ s_i) {
    __shared__ __align__(16) bf16 BrT[256][40];   // [col][k-chunk]
    __shared__ __align__(16) bf16 BiT[256][40];

    const int m    = blockIdx.x;
    const int tid  = threadIdx.x;
    const int lane = tid & 31;
    const int wid  = tid >> 5;
    const int c0   = wid * 32;     // each wave: cols [c0, c0+32)
    const int bp   = lane & 15;    // batch row of A fragment
    const int hl   = lane >> 4;

    v8f ar0 = {}, ar1 = {}, ai0 = {}, ai1 = {};

    for (int k0 = 0; k0 < 256; k0 += 32) {
        // stage: thread t copies 32 contiguous k of column n = t (64B each)
        {
            const bf16* pR = WrT + (((long long)m) << 16) + (long long)tid * 256 + k0;
            const bf16* pI = WiT + (((long long)m) << 16) + (long long)tid * 256 + k0;
            #pragma unroll
            for (int q = 0; q < 4; ++q) {
                *(uint4*)&BrT[tid][q * 8] = *(const uint4*)(pR + q * 8);
                *(uint4*)&BiT[tid][q * 8] = *(const uint4*)(pI + q * 8);
            }
        }
        __syncthreads();

        // A fragments: two contiguous 8-element runs per lane
        FragBF a_r, a_i, a_in;
        if (bp < 8) {
            const bf16* pr = ut + (long long)bp * (128 * 256) + m * 256 + k0 + (hl << 3);
            a_r.q[0] = *(const uint4*)(pr);
            a_r.q[1] = *(const uint4*)(pr + 16);
            const bf16* pi = pr + 64 * 256;
            a_i.q[0] = *(const uint4*)(pi);
            a_i.q[1] = *(const uint4*)(pi + 16);
        } else {
            a_r.q[0] = make_uint4(0, 0, 0, 0); a_r.q[1] = make_uint4(0, 0, 0, 0);
            a_i.q[0] = make_uint4(0, 0, 0, 0); a_i.q[1] = make_uint4(0, 0, 0, 0);
        }
        // -Ai via packed sign-bit flip
        #pragma unroll
        for (int j = 0; j < 2; ++j) {
            a_in.q[j].x = a_i.q[j].x ^ 0x80008000u;
            a_in.q[j].y = a_i.q[j].y ^ 0x80008000u;
            a_in.q[j].z = a_i.q[j].z ^ 0x80008000u;
            a_in.q[j].w = a_i.q[j].w ^ 0x80008000u;
        }

        // B fragments: two 8-element k runs per lane (k = half*8, 16 + half*8) at fixed column
        FragBF br0, br1, bi0, bi1;
        {
            const bf16* p0 = &BrT[c0 + (lane & 15)][hl << 3];
            br0.q[0] = *(const uint4*)(p0); br0.q[1] = *(const uint4*)(p0 + 16);
            const bf16* p1 = &BrT[c0 + 16 + (lane & 15)][hl << 3];
            br1.q[0] = *(const uint4*)(p1); br1.q[1] = *(const uint4*)(p1 + 16);
            const bf16* p2 = &BiT[c0 + (lane & 15)][hl << 3];
            bi0.q[0] = *(const uint4*)(p2); bi0.q[1] = *(const uint4*)(p2 + 16);
            const bf16* p3 = &BiT[c0 + 16 + (lane & 15)][hl << 3];
            bi1.q[0] = *(const uint4*)(p3); bi1.q[1] = *(const uint4*)(p3 + 16);
        }

        // Wu_r = Ar@Wr - Ai@Wi ;  Wu_i = Ar@Wi + Ai@Wr
        ar0 = __builtin_amdgcn_wmma_f32_16x16x32_bf16(false, a_r.v,  false, br0.v, (short)0, ar0, false, false);
        ar0 = __builtin_amdgcn_wmma_f32_16x16x32_bf16(false, a_in.v, false, bi0.v, (short)0, ar0, false, false);
        ai0 = __builtin_amdgcn_wmma_f32_16x16x32_bf16(false, a_r.v,  false, bi0.v, (short)0, ai0, false, false);
        ai0 = __builtin_amdgcn_wmma_f32_16x16x32_bf16(false, a_i.v,  false, br0.v, (short)0, ai0, false, false);
        ar1 = __builtin_amdgcn_wmma_f32_16x16x32_bf16(false, a_r.v,  false, br1.v, (short)0, ar1, false, false);
        ar1 = __builtin_amdgcn_wmma_f32_16x16x32_bf16(false, a_in.v, false, bi1.v, (short)0, ar1, false, false);
        ai1 = __builtin_amdgcn_wmma_f32_16x16x32_bf16(false, a_r.v,  false, bi1.v, (short)0, ai1, false, false);
        ai1 = __builtin_amdgcn_wmma_f32_16x16x32_bf16(false, a_i.v,  false, br1.v, (short)0, ai1, false, false);
        __syncthreads();
    }

    const int n = lane & 15;
    #pragma unroll
    for (int r = 0; r < 8; ++r) {
        int b = r + (hl << 3);
        if (b < 8) {
            long long o = (long long)b * (64 * 256) + (long long)m * 256;
            s_r[o + c0 + n]      = gelu_exact(ar0[r]);
            s_r[o + c0 + 16 + n] = gelu_exact(ar1[r]);
            s_i[o + c0 + n]      = gelu_exact(ai0[r]);
            s_i[o + c0 + 16 + n] = gelu_exact(ai1[r]);
        }
    }
}

// ------------------------------- launcher ----------------------------------

extern "C" void kernel_launch(void* const* d_in, const int* in_sizes, int n_in,
                              void* d_out, int out_size, void* d_ws, size_t ws_size,
                              hipStream_t stream) {
    (void)in_sizes; (void)n_in; (void)out_size; (void)ws_size;

    const float* u      = (const float*)d_in[0];  // [8][8192][256]
    const float* Wr     = (const float*)d_in[1];  // [64][256][256]
    const float* Wi     = (const float*)d_in[2];
    const float* skip_w = (const float*)d_in[3];  // [256][256]
    const float* skip_b = (const float*)d_in[4];  // [256]
    const float* w1     = (const float*)d_in[5];  // [128][256]
    const float* b1     = (const float*)d_in[6];  // [128]
    const float* w2     = (const float*)d_in[7];  // [256][128]
    const float* b2     = (const float*)d_in[8];  // [256]
    float*       out    = (float*)d_out;          // [8][8192][256]

    const int Bn = 8, S = 8192, H = 256, Md = 64, Hm = 128;
    const long long NR = (long long)Bn * S;       // 65536 rows

    size_t off = 0;
    char* wsb = (char*)d_ws;
    auto alloc = [&](size_t bytes) -> void* {
        off = (off + 255) & ~((size_t)255);
        void* p = (void*)(wsb + off);
        off += bytes;
        return p;
    };

    bf16*  u_bf   = (bf16*) alloc((size_t)NR * H * 2);            // 33.5 MB
    bf16*  WrT_bf = (bf16*) alloc((size_t)Md * H * H * 2);        //  8.4 MB
    bf16*  WiT_bf = (bf16*) alloc((size_t)Md * H * H * 2);        //  8.4 MB
    bf16*  swT    = (bf16*) alloc((size_t)H * H * 2);
    bf16*  w1T    = (bf16*) alloc((size_t)H * Hm * 2);
    bf16*  w2T    = (bf16*) alloc((size_t)Hm * H * 2);
    bf16*  fB     = (bf16*) alloc((size_t)128 * S * 2);           //  2.1 MB
    bf16*  iB     = (bf16*) alloc((size_t)S * 128 * 2);           //  2.1 MB
    float* ut_f   = (float*)alloc((size_t)Bn * 128 * H * 4);      //  1.0 MB
    bf16*  ut_b   = (bf16*) alloc((size_t)Bn * 128 * H * 2);
    float* s_r    = (float*)alloc((size_t)Bn * Md * H * 4);
    float* s_i    = (float*)alloc((size_t)Bn * Md * H * 4);
    bf16*  conv_b = (bf16*) alloc((size_t)Bn * 128 * H * 2);
    bf16*  y1     = (bf16*) alloc((size_t)NR * H * 2);            // 33.5 MB
    bf16*  act    = (bf16*) alloc((size_t)NR * H * 2);            // 33.5 MB
    bf16*  h_bf   = (bf16*) alloc((size_t)NR * Hm * 2);           // 16.8 MB
    // total workspace ~141 MB

    // --- precision casts / transposes / basis tables ---
    {
        int n4 = (int)(NR * H / 4);
        k_cvt_bf16_v4<<<dim3((n4 + 255) / 256), dim3(256), 0, stream>>>(u, u_bf, n4);
    }
    k_transpose_cvt_bf16<<<dim3(8, 8, 64), dim3(256), 0, stream>>>(Wr, WrT_bf, H, H);
    k_transpose_cvt_bf16<<<dim3(8, 8, 64), dim3(256), 0, stream>>>(Wi, WiT_bf, H, H);
    k_transpose_cvt_bf16<<<dim3(8, 8, 1),  dim3(256), 0, stream>>>(skip_w, swT, H, H);
    k_transpose_cvt_bf16<<<dim3(8, 4, 1),  dim3(256), 0, stream>>>(w1, w1T, Hm, H);
    k_transpose_cvt_bf16<<<dim3(4, 8, 1),  dim3(256), 0, stream>>>(w2, w2T, H, Hm);
    k_gen_fwd_basis<<<dim3(128 * S / 256), dim3(256), 0, stream>>>(fB);
    k_gen_inv_basis<<<dim3(S * 128 / 256), dim3(256), 0, stream>>>(iB);

    // --- 1) forward DFT: ut_cat[b] (128x256) = fB (128x8192) @ u_bf[b] ---
    k_gemm_bf16<false, false, false, true, true>
        <<<dim3(H / 64, 128 / 32, Bn), dim3(256), 0, stream>>>(
        fB, u_bf, ut_f, ut_b, nullptr, nullptr,
        /*K*/ S, /*lda*/ S, /*ldb*/ H, /*ldc*/ H,
        0LL, (long long)S * H, (long long)128 * H);

    // --- 2) per-mode complex mix + GELU -> s ---
    k_modemix<<<dim3(Md), dim3(256), 0, stream>>>(ut_b, WrT_bf, WiT_bf, s_r, s_i);

    // --- 3) 64-point circular convolution over modes ---
    k_circconv<<<dim3(Bn * Md * H / 256), dim3(256), 0, stream>>>(ut_f, s_r, s_i, conv_b);

    // --- 4) inverse DFT: y1[b] (8192x256) = iB (8192x128) @ conv_b[b] ---
    k_gemm_bf16<false, false, false, false, true>
        <<<dim3(H / 64, S / 32, Bn), dim3(256), 0, stream>>>(
        iB, conv_b, nullptr, y1, nullptr, nullptr,
        /*K*/ 128, /*lda*/ 128, /*ldb*/ H, /*ldc*/ H,
        0LL, (long long)128 * H, (long long)S * H);

    // --- 5) act = gelu( u @ skip_w.T + skip_b + y1 ) ---
    k_gemm_bf16<true, true, true, false, true>
        <<<dim3(H / 64, (int)(NR / 32), 1), dim3(256), 0, stream>>>(
        u_bf, swT, nullptr, act, y1, skip_b,
        /*K*/ H, /*lda*/ H, /*ldb*/ H, /*ldc*/ H, 0LL, 0LL, 0LL);

    // --- 6) h = gelu( act @ mlp_w1.T + b1 ) ---
    k_gemm_bf16<false, true, true, false, true>
        <<<dim3(Hm / 64, (int)(NR / 32), 1), dim3(256), 0, stream>>>(
        act, w1T, nullptr, h_bf, nullptr, b1,
        /*K*/ H, /*lda*/ H, /*ldb*/ Hm, /*ldc*/ Hm, 0LL, 0LL, 0LL);

    // --- 7) out = h @ mlp_w2.T + b2  (f32 to d_out) ---
    k_gemm_bf16<false, true, false, true, false>
        <<<dim3(H / 64, (int)(NR / 32), 1), dim3(256), 0, stream>>>(
        h_bf, w2T, out, nullptr, nullptr, b2,
        /*K*/ Hm, /*lda*/ Hm, /*ldb*/ H, /*ldc*/ H, 0LL, 0LL, 0LL);
}
